// Quantizer_19000935317796
// MI455X (gfx1250) — compile-verified, matrix operand order checked by reading, not run on hardware
//
#include <hip/hip_runtime.h>
#include <hip/hip_bf16.h>

// VQ nearest-neighbor quantizer for MI455X (gfx1250, wave32).
//   z:        [32, 64, 32, 32] f32   (B=32, C=64, H=W=32; N = B*H*W = 32768)
//   codebook: [1024, 64] f32         (K=1024)
// Outputs (concatenated in d_out): codes [32,64,32,32] f32 (2097152 floats),
// then indices [32,32,32] int32 (32768).
//
// argmin_k (||c_k||^2 - 2 x.c_k): the -2 x.c term is a 32768x64 @ 64x1024 GEMM
// done with v_wmma_f32_16x16x32_bf16 (f32 accumulate), C=64 -> 2 chained WMMAs
// per 16x16 tile. B fragments + codebook norms live in LDS (132 KB/WGP),
// filled via async global->LDS copies; B loads are software-pipelined one
// tile ahead to overlap ds_load latency with WMMA + argmin VALU.

typedef __attribute__((ext_vector_type(16))) __bf16 v16bf;
typedef __attribute__((ext_vector_type(8)))  float  v8f;

#define N_ROWS   32768
#define N_CODES  1024
#define C_DIM    64
#define HW       1024
#define CODES_ELEMS (32 * 64 * 1024)   // 2097152
#define BPACK_BYTES 131072             // 2*64*32*16 bf16
#define LDS_BYTES   135168             // + 1024 f32 norms

// WMMA bf16 A/B fragment K-index for element e of the 16-element per-lane
// vector (ISA 7.12.2, 16-bit A 16x32 layout; B conformal):
__device__ __forceinline__ int frag_k(int e, int half) {
    return e + (half << 3) + ((e >= 8) ? 8 : 0);
}

// ---------------------------------------------------------------------------
// Prep 1: repack codebook into per-lane WMMA B-fragment order, bf16.
// bpack[((kt*64 + nt)*32 + lane)*16 + e]
//   = bf16( codebook[(nt*16 + lane%16)*64 + kt*32 + K(e, lane/16)] )
__global__ __launch_bounds__(256) void vq_pack(const float* __restrict__ cb,
                                               __bf16* __restrict__ bpack) {
    int t = blockIdx.x * 256 + threadIdx.x;      // 0 .. 4095
    int lane  = t & 31;
    int nt    = (t >> 5) & 63;
    int kt    = t >> 11;
    int ncol  = lane & 15;
    int khalf = lane >> 4;
    const float* src = cb + ((nt << 4) + ncol) * C_DIM + (kt << 5);
    __bf16* dst = bpack + ((size_t)t << 4);
#pragma unroll
    for (int e = 0; e < 16; ++e)
        dst[e] = (__bf16)src[frag_k(e, khalf)];
}

// Prep 2: ||c_k||^2, packed per-lane-contiguous: cnormP[(k%16)*64 + k/16]
// so each lane reads 4 consecutive tiles' norms with one b128.
__global__ __launch_bounds__(256) void vq_norm(const float* __restrict__ cb,
                                               float* __restrict__ cnormP) {
    int k = blockIdx.x * 256 + threadIdx.x;      // 0 .. 1023
    const float* row = cb + k * C_DIM;
    float s = 0.f;
#pragma unroll
    for (int c = 0; c < C_DIM; ++c) s = __builtin_fmaf(row[c], row[c], s);
    cnormP[(k & 15) * 64 + (k >> 4)] = s;
}

// ---------------------------------------------------------------------------
// Main: one wave per 16-row strip; full codebook (B frags + norms) in LDS.
__global__ __launch_bounds__(256) void vq_main(const float* __restrict__ z,
                                               const char* __restrict__ wsrc,
                                               int* __restrict__ idx_out) {
    extern __shared__ __align__(16) char smem[];
    __bf16* sB = (__bf16*)smem;                       // 128 KB B fragments
    float*  sN = (float*)(smem + BPACK_BYTES);        //   4 KB packed norms

    int t = threadIdx.x;

    // Async global -> LDS fill (bypasses VGPRs): 135168 B = 8448 x b128,
    // 33 chunks per thread. GLOBAL_LOAD_ASYNC_TO_LDS_B128, tracked by ASYNCcnt.
    for (int i = t; i < (LDS_BYTES / 16); i += 256) {
        unsigned lds_off = (unsigned)(i * 16);
        unsigned long long gaddr = (unsigned long long)(wsrc + (size_t)i * 16);
        asm volatile("global_load_async_to_lds_b128 %0, %1, off"
                     :: "v"(lds_off), "v"(gaddr) : "memory");
    }
    asm volatile("s_wait_asynccnt 0" ::: "memory");
    __syncthreads();

    int lane = t & 31;
    int half = lane >> 4;            // 0: D rows v, 1: D rows v+8
    int mrow = lane & 15;            // A row within strip / D column in tile

    int strip = blockIdx.x * 8 + (t >> 5);   // 0 .. 2047
    int n0 = strip << 4;
    int b  = n0 >> 10;                       // 16 | HW -> strip stays in one b
    int hw = n0 & (HW - 1);

    // A fragments: rows n0..n0+15. x[n][c] = z[b*C*HW + c*HW + hw];
    // lanes 0..15 hit 16 consecutive floats per element -> coalesced.
    v16bf a0, a1;
    const float* zb = z + (size_t)b * (C_DIM * HW) + hw + mrow;
#pragma unroll
    for (int e = 0; e < 16; ++e) {
        int K = frag_k(e, half);
        a0[e] = (__bf16)zb[(size_t)K * HW];
        a1[e] = (__bf16)zb[(size_t)(K + 32) * HW];
    }

    float minv[8];
    int   mini[8];
#pragma unroll
    for (int v = 0; v < 8; ++v) { minv[v] = 3.4e38f; mini[v] = 0; }

#define LDB(kt, nt) (*(const v16bf*)(sB + ((((kt) * 64 + (nt)) * 32 + lane) << 4)))

    // Software pipeline: B fragments for tile nt+1 issue before tile nt's
    // WMMAs + argmin VALU, hiding ds_load latency at 2 waves/SIMD occupancy.
    v16bf b0 = LDB(0, 0);
    v16bf b1 = LDB(1, 0);
    const float4* cnp = (const float4*)sN + (mrow << 4);

    for (int nt4 = 0; nt4 < 16; ++nt4) {
        float4 cn4 = cnp[nt4];
#pragma unroll
        for (int j = 0; j < 4; ++j) {
            int nt  = (nt4 << 2) + j;
            int ntn = (nt + 1) & 63;         // branchless wrap (tile 0 reload ok)
            v16bf nb0 = LDB(0, ntn);
            v16bf nb1 = LDB(1, ntn);

            v8f acc = {};
            acc = __builtin_amdgcn_wmma_f32_16x16x32_bf16(
                      false, a0, false, b0, (short)0, acc, false, false);
            acc = __builtin_amdgcn_wmma_f32_16x16x32_bf16(
                      false, a1, false, b1, (short)0, acc, false, false);

            float cn  = (j == 0) ? cn4.x : (j == 1) ? cn4.y : (j == 2) ? cn4.z : cn4.w;
            int   col = (nt << 4) + mrow;
#pragma unroll
            for (int v = 0; v < 8; ++v) {
                float d = __builtin_fmaf(-2.0f, acc[v], cn);
                if (d < minv[v]) { minv[v] = d; mini[v] = col; }
            }
            b0 = nb0;
            b1 = nb1;
        }
    }
#undef LDB

    // Branchless butterfly argmin across the 16 lanes holding one tile-row
    // each (xor masks < 16 keep the two 16-lane halves separate).
#pragma unroll
    for (int off = 1; off < 16; off <<= 1) {
#pragma unroll
        for (int v = 0; v < 8; ++v) {
            float ov = __shfl_xor(minv[v], off, 32);
            int   oi = __shfl_xor(mini[v], off, 32);
            int take = (int)(ov < minv[v]) |
                       ((int)(ov == minv[v]) & (int)(oi < mini[v]));
            minv[v] = take ? ov : minv[v];
            mini[v] = take ? oi : mini[v];
        }
    }

    if (mrow == 0) {                 // lane 0 -> rows 0..7, lane 16 -> rows 8..15
#pragma unroll
        for (int v = 0; v < 8; ++v)
            idx_out[n0 + v + (half << 3)] = mini[v];
    }
}

// ---------------------------------------------------------------------------
// codes[b,c,h,w] = codebook[idx[b,h,w]][c]; output-coalesced, cb hits L2.
__global__ __launch_bounds__(256) void vq_codes(const float* __restrict__ cb,
                                                const int* __restrict__ idx,
                                                float* __restrict__ codes) {
    int o  = blockIdx.x * 256 + threadIdx.x;   // 0 .. 2097151
    int b  = o >> 16;
    int c  = (o >> 10) & 63;
    int hw = o & (HW - 1);
    int n  = (b << 10) | hw;
    int k  = idx[n];
    codes[o] = cb[k * C_DIM + c];
}

// ---------------------------------------------------------------------------
extern "C" void kernel_launch(void* const* d_in, const int* in_sizes, int n_in,
                              void* d_out, int out_size, void* d_ws, size_t ws_size,
                              hipStream_t stream) {
    (void)in_sizes; (void)n_in; (void)out_size; (void)ws_size;

    const float* z  = (const float*)d_in[0];
    const float* cb = (const float*)d_in[1];

    float* codes   = (float*)d_out;
    int*   idx_out = (int*)((float*)d_out + CODES_ELEMS);

    __bf16* bpack = (__bf16*)d_ws;                          // 128 KB
    float*  cnorm = (float*)((char*)d_ws + BPACK_BYTES);    //   4 KB (packed)

    vq_pack <<<16,   256, 0, stream>>>(cb, bpack);
    vq_norm <<<4,    256, 0, stream>>>(cb, cnorm);
    vq_main <<<256,  256, LDS_BYTES, stream>>>(z, (const char*)d_ws, idx_out);
    vq_codes<<<8192, 256, 0, stream>>>(cb, idx_out, codes);
}
